// RelationshipClassifier_13864154431828
// MI455X (gfx1250) — matrix operand order repaired and checked by reading, not compiled, not run on hardware
//
#include <hip/hip_runtime.h>
#include <hip/hip_bf16.h>
#include <math.h>

// ---------------------------------------------------------------------------
// Problem dims (from reference)
// ---------------------------------------------------------------------------
#define BATCH   16384
#define EDIM    512
#define HDIM    1024
#define PDIM    256
#define CDIM    7
#define DIN     1026
#define DPAD    1056   // 33 K-tiles of 32

typedef __attribute__((ext_vector_type(16))) __bf16 v16bf;
typedef __attribute__((ext_vector_type(8)))  float  v8f;

__device__ __forceinline__ unsigned short f32_to_bf16_rne(float f) {
    unsigned int u = __float_as_uint(f);
    unsigned int r = u + 0x7FFFu + ((u >> 16) & 1u);
    return (unsigned short)(r >> 16);
}

// ---------------------------------------------------------------------------
// Kernel 1: L2-normalize x1,x2 and build padded combined [B, 1056] f32
// One block (256 threads) per row.
// ---------------------------------------------------------------------------
__global__ __launch_bounds__(256) void prep_combined(
    const float* __restrict__ x1, const float* __restrict__ x2,
    const float* __restrict__ gf, float* __restrict__ comb)
{
    const int row  = blockIdx.x;
    const int lane = threadIdx.x & 31;
    const int wave = threadIdx.x >> 5;
    __shared__ float red1[8], red2[8], inv[2];

    float s1 = 0.f, s2 = 0.f;
    for (int i = threadIdx.x; i < EDIM; i += 256) {
        float a = x1[(size_t)row * EDIM + i]; s1 += a * a;
        float b = x2[(size_t)row * EDIM + i]; s2 += b * b;
    }
    #pragma unroll
    for (int off = 1; off < 32; off <<= 1) {
        s1 += __shfl_xor(s1, off, 32);
        s2 += __shfl_xor(s2, off, 32);
    }
    if (lane == 0) { red1[wave] = s1; red2[wave] = s2; }
    __syncthreads();
    if (threadIdx.x == 0) {
        float t1 = 0.f, t2 = 0.f;
        for (int w = 0; w < 8; ++w) { t1 += red1[w]; t2 += red2[w]; }
        inv[0] = 1.f / fmaxf(sqrtf(t1), 1e-12f);
        inv[1] = 1.f / fmaxf(sqrtf(t2), 1e-12f);
    }
    __syncthreads();
    const float i1 = inv[0], i2 = inv[1];
    const size_t base = (size_t)row * DPAD;
    for (int i = threadIdx.x; i < EDIM; i += 256) {
        comb[base + i]        = x1[(size_t)row * EDIM + i] * i1;
        comb[base + EDIM + i] = x2[(size_t)row * EDIM + i] * i2;
    }
    if (threadIdx.x < 2)
        comb[base + 2 * EDIM + threadIdx.x] = gf[(size_t)row * 2 + threadIdx.x];
    else if (threadIdx.x < 32)
        comb[base + 2 * EDIM + threadIdx.x] = 0.f;   // zero pad 1026..1055
}

// ---------------------------------------------------------------------------
// Kernel 2: pack f32 weights [K_real, N] into bf16 WMMA B-fragment layout.
// Per (kt, nt) 32x16 tile: 32 lanes x 16 bf16 contiguous (1 KB).
// B-fragment k order used here (contiguous, same as A): lanes 0-15 hold K=0..15 of
// column n=lane, lanes 16-31 hold K=16..31 of column n=lane-16; element
// index e within the lane's v16bf = k % 16.
// ---------------------------------------------------------------------------
__global__ __launch_bounds__(256) void pack_weights(
    const float* __restrict__ W, unsigned short* __restrict__ Wp,
    int K_real, int N, int total)
{
    int idx = blockIdx.x * 256 + threadIdx.x;
    if (idx >= total) return;
    const int e    = idx & 15;
    const int lane = (idx >> 4) & 31;
    const int tile = idx >> 9;
    const int NT   = N >> 4;
    const int nt   = tile % NT;
    const int kt   = tile / NT;
    const int n = (nt << 4) + (lane & 15);
    const int k = (kt << 5) + ((lane >> 4) << 4) + e;
    float v = (k < K_real) ? W[(size_t)k * N + n] : 0.f;
    Wp[idx] = f32_to_bf16_rne(v);
}

// ---------------------------------------------------------------------------
// Kernel 3: fused  out = [resid +] ReLU(LN(A @ W + bias) * gamma + beta)
// Block = 16 rows x N_COLS cols, 8 wave32s, wave owns N_COLS/128 col-tiles.
// Double-buffered LDS A staging (1 barrier / K-step).  B-fragment stream is
// software-pipelined at half-tile-group granularity (two live groups bA/bB)
// so ~8 global_load_b128 stay in flight past every WMMA group and across the
// barrier, instead of s_wait_loadcnt 0x0 before each WMMA.
// A-fragment layout: lane = m + 16*(k/16); elem e = k%16.
// C/D layout: VGPR r, lanes 0-15 -> (M=r, N=lane); lanes 16-31 -> (M=r+8).
// ---------------------------------------------------------------------------
template<int N_COLS, int K_TILES, bool RESIDUAL>
__global__ __launch_bounds__(256) void gemm_ln_relu(
    const float* __restrict__ A, const unsigned short* __restrict__ Wp,
    const float* __restrict__ bias, const float* __restrict__ gamma,
    const float* __restrict__ beta, const float* __restrict__ resid,
    float* __restrict__ out, int A_stride)
{
    constexpr int NT   = N_COLS / 16;
    constexpr int TPW  = NT / 8;                // col-tiles per wave
    constexpr int HTPW = (TPW >= 2) ? TPW / 2 : 1;  // pipelined half group
    __shared__ __align__(32) unsigned short ldsA[2][16 * 32];
    __shared__ float ldsSum[128], ldsSq[128], ldsMean[16], ldsRstd[16];

    const int lane    = threadIdx.x & 31;
    const int wave    = threadIdx.x >> 5;
    const int rowBase = blockIdx.x << 4;

    const v8f vzero = {0.f, 0.f, 0.f, 0.f, 0.f, 0.f, 0.f, 0.f};
    v8f acc[TPW];
    #pragma unroll
    for (int t = 0; t < TPW; ++t) acc[t] = vzero;

    // stage A tile kt into LDS buffer buf (f32 load + bf16 convert + pack)
    auto stageA = [&](int kt, int buf) {
        #pragma unroll
        for (int i = threadIdx.x; i < 512; i += 256) {
            const int m = i >> 5, k = i & 31;
            float f = A[(size_t)(rowBase + m) * A_stride + (kt << 5) + k];
            const int l = m + ((k >> 4) << 4);
            const int e = k & 15;
            ldsA[buf][(l << 4) + e] = f32_to_bf16_rne(f);
        }
    };

    // load one half-group of B fragments for K-tile kt
    const unsigned short* wlane = Wp + (lane << 4);
    auto loadB = [&](int kt, int half, v16bf* dst) {
        const size_t base = (((size_t)kt * NT) + wave * TPW + half * HTPW) << 9;
        #pragma unroll
        for (int t = 0; t < HTPW; ++t)
            dst[t] = *reinterpret_cast<const v16bf*>(
                         &wlane[base + ((size_t)t << 9)]);
    };

    v16bf bA[HTPW], bB[HTPW];
    loadB(0, 0, bA);
    stageA(0, 0);
    __syncthreads();

    for (int kt = 0; kt < K_TILES; ++kt) {
        const int buf = kt & 1;
        v16bf a = *reinterpret_cast<const v16bf*>(&ldsA[buf][lane << 4]);
        // overlap next A tile staging with this step's B loads + WMMAs
        if (kt + 1 < K_TILES) stageA(kt + 1, buf ^ 1);
        if (kt + 2 < K_TILES) {
            // near-cache prefetch 2 K-tiles ahead of the weight stream
            __builtin_prefetch(
                wlane + ((((size_t)(kt + 2) * NT) + wave * TPW) << 9), 0, 3);
        }
        if (TPW >= 2) {
            loadB(kt, 1, bB);                    // second half of this tile
            #pragma unroll
            for (int t = 0; t < HTPW; ++t)
                acc[t] = __builtin_amdgcn_wmma_f32_16x16x32_bf16(
                             false, a, false, bA[t], (short)0, acc[t], false, false);
            if (kt + 1 < K_TILES) loadB(kt + 1, 0, bA);   // first half of next
            #pragma unroll
            for (int t = 0; t < HTPW; ++t)
                acc[HTPW + t] = __builtin_amdgcn_wmma_f32_16x16x32_bf16(
                             false, a, false, bB[t], (short)0, acc[HTPW + t], false, false);
        } else {
            // TPW==1: single tile per wave, simple prefetch-next pipeline
            acc[0] = __builtin_amdgcn_wmma_f32_16x16x32_bf16(
                         false, a, false, bA[0], (short)0, acc[0], false, false);
            if (kt + 1 < K_TILES) loadB(kt + 1, 0, bA);
        }
        __syncthreads();
    }

    // bias add + per-row partial sums (rows: r for lanes 0-15, r+8 for 16-31)
    float srow[8], sqrow[8];
    #pragma unroll
    for (int r = 0; r < 8; ++r) { srow[r] = 0.f; sqrow[r] = 0.f; }
    #pragma unroll
    for (int t = 0; t < TPW; ++t) {
        const int n = ((wave * TPW + t) << 4) + (lane & 15);
        const float bn = bias[n];
        #pragma unroll
        for (int r = 0; r < 8; ++r) {
            float v = acc[t][r] + bn;
            acc[t][r] = v;
            srow[r]  += v;
            sqrow[r] += v * v;
        }
    }
    // reduce across the 16 lanes of each half-wave (halves are distinct rows)
    #pragma unroll
    for (int off = 1; off < 16; off <<= 1) {
        #pragma unroll
        for (int r = 0; r < 8; ++r) {
            srow[r]  += __shfl_xor(srow[r],  off, 32);
            sqrow[r] += __shfl_xor(sqrow[r], off, 32);
        }
    }
    if (lane == 0 || lane == 16) {
        const int ro = (lane >> 4) << 3;
        #pragma unroll
        for (int r = 0; r < 8; ++r) {
            ldsSum[(wave << 4) + ro + r] = srow[r];
            ldsSq [(wave << 4) + ro + r] = sqrow[r];
        }
    }
    __syncthreads();
    if (threadIdx.x < 16) {
        float s = 0.f, q = 0.f;
        for (int w = 0; w < 8; ++w) {
            s += ldsSum[(w << 4) + threadIdx.x];
            q += ldsSq [(w << 4) + threadIdx.x];
        }
        const float mean = s * (1.0f / N_COLS);
        const float var  = q * (1.0f / N_COLS) - mean * mean;
        ldsMean[threadIdx.x] = mean;
        ldsRstd[threadIdx.x] = rsqrtf(var + 1e-5f);
    }
    __syncthreads();

    // LN * gamma + beta, ReLU, optional residual, store f32
    const int halfs = lane >> 4;
    const int ncol  = lane & 15;
    #pragma unroll
    for (int t = 0; t < TPW; ++t) {
        const int n = ((wave * TPW + t) << 4) + ncol;
        const float gn = gamma[n], btn = beta[n];
        #pragma unroll
        for (int r = 0; r < 8; ++r) {
            const int rowl = r + (halfs << 3);
            float v = (acc[t][r] - ldsMean[rowl]) * ldsRstd[rowl] * gn + btn;
            v = fmaxf(v, 0.f);
            const size_t o = (size_t)(rowBase + rowl) * N_COLS + n;
            if (RESIDUAL) v += resid[o];
            out[o] = v;
        }
    }
}

// ---------------------------------------------------------------------------
// Kernel 4: classifier [B,256]@[256,7] + bias + gender mask  (tiny, VALU)
// ---------------------------------------------------------------------------
__global__ __launch_bounds__(256) void classifier_mask(
    const float* __restrict__ pf, const float* __restrict__ Wc,
    const float* __restrict__ bc, const float* __restrict__ gf,
    float* __restrict__ outp)
{
    const int idx = blockIdx.x * 256 + threadIdx.x;
    if (idx >= BATCH * CDIM) return;
    const int row = idx / CDIM;
    const int c   = idx - row * CDIM;
    float acc = bc[c];
    const float* prow = pf + (size_t)row * PDIM;
    for (int p = 0; p < PDIM; ++p)
        acc = fmaf(prow[p], Wc[(size_t)p * CDIM + c], acc);
    const int g1 = (int)gf[(size_t)row * 2 + 0];
    const int g2 = (int)gf[(size_t)row * 2 + 1];
    bool valid;
    if (g1 == 0 && g2 == 0)      valid = (c == 1 || c == 4);
    else if (g1 == 1 && g2 == 1) valid = (c == 2 || c == 5);
    else                         valid = (c == 0 || c == 3 || c == 6);
    outp[idx] = acc + (valid ? 0.f : __int_as_float(0xff800000u)); // + (-inf)
}

// ---------------------------------------------------------------------------
// Host launcher
// ---------------------------------------------------------------------------
static inline size_t align256(size_t x) { return (x + 255) & ~(size_t)255; }

extern "C" void kernel_launch(void* const* d_in, const int* in_sizes, int n_in,
                              void* d_out, int out_size, void* d_ws, size_t ws_size,
                              hipStream_t stream) {
    const float* x1    = (const float*)d_in[0];
    const float* x2    = (const float*)d_in[1];
    const float* gf    = (const float*)d_in[2];
    const float* W_in  = (const float*)d_in[3];
    const float* b_in  = (const float*)d_in[4];
    const float* g_in  = (const float*)d_in[5];
    const float* be_in = (const float*)d_in[6];
    const float* W_r1  = (const float*)d_in[7];
    const float* b_r1  = (const float*)d_in[8];
    const float* g_r1  = (const float*)d_in[9];
    const float* be_r1 = (const float*)d_in[10];
    const float* W_r2  = (const float*)d_in[11];
    const float* b_r2  = (const float*)d_in[12];
    const float* g_r2  = (const float*)d_in[13];
    const float* be_r2 = (const float*)d_in[14];
    const float* W_f   = (const float*)d_in[15];
    const float* b_f   = (const float*)d_in[16];
    const float* g_f   = (const float*)d_in[17];
    const float* be_f  = (const float*)d_in[18];
    const float* W_c   = (const float*)d_in[19];
    const float* b_c   = (const float*)d_in[20];

    char* ws = (char*)d_ws;
    size_t off = 0;
    float* comb   = (float*)(ws + off); off = align256(off + (size_t)BATCH * DPAD * 4);
    float* featsA = (float*)(ws + off); off = align256(off + (size_t)BATCH * HDIM * 4);
    float* featsB = (float*)(ws + off); off = align256(off + (size_t)BATCH * HDIM * 4);
    float* pfeats = (float*)(ws + off); off = align256(off + (size_t)BATCH * PDIM * 4);
    const int packIn = 33 * (HDIM / 16) * 512;       // W_in padded K=1056
    const int packR  = 32 * (HDIM / 16) * 512;       // W_r1 / W_r2
    const int packF  = 32 * (PDIM / 16) * 512;       // W_f
    unsigned short* WpIn = (unsigned short*)(ws + off); off = align256(off + (size_t)packIn * 2);
    unsigned short* WpR1 = (unsigned short*)(ws + off); off = align256(off + (size_t)packR * 2);
    unsigned short* WpR2 = (unsigned short*)(ws + off); off = align256(off + (size_t)packR * 2);
    unsigned short* WpF  = (unsigned short*)(ws + off); off = align256(off + (size_t)packF * 2);

    // Stage 0: activations + weight packing (independent, same stream)
    prep_combined<<<BATCH, 256, 0, stream>>>(x1, x2, gf, comb);
    pack_weights<<<(packIn + 255) / 256, 256, 0, stream>>>(W_in, WpIn, DIN,  HDIM, packIn);
    pack_weights<<<(packR  + 255) / 256, 256, 0, stream>>>(W_r1, WpR1, HDIM, HDIM, packR);
    pack_weights<<<(packR  + 255) / 256, 256, 0, stream>>>(W_r2, WpR2, HDIM, HDIM, packR);
    pack_weights<<<(packF  + 255) / 256, 256, 0, stream>>>(W_f,  WpF,  HDIM, PDIM, packF);

    const int gemmBlocks = BATCH / 16;
    // Stage 1: input_proj
    gemm_ln_relu<HDIM, 33, false><<<gemmBlocks, 256, 0, stream>>>(
        comb, WpIn, b_in, g_in, be_in, nullptr, featsA, DPAD);
    // Stage 2: residual block 1  (featsB = featsA + ReLU(LN(featsA@W)))
    gemm_ln_relu<HDIM, 32, true><<<gemmBlocks, 256, 0, stream>>>(
        featsA, WpR1, b_r1, g_r1, be_r1, featsA, featsB, HDIM);
    // Stage 3: residual block 2
    gemm_ln_relu<HDIM, 32, true><<<gemmBlocks, 256, 0, stream>>>(
        featsB, WpR2, b_r2, g_r2, be_r2, featsB, featsA, HDIM);
    // Stage 4: final_proj
    gemm_ln_relu<PDIM, 32, false><<<gemmBlocks, 256, 0, stream>>>(
        featsA, WpF, b_f, g_f, be_f, nullptr, pfeats, HDIM);
    // Stage 5: classifier + mask
    classifier_mask<<<(BATCH * CDIM + 255) / 256, 256, 0, stream>>>(
        pfeats, W_c, b_c, gf, (float*)d_out);
}
